// ImplicitLinear_43112881717880
// MI455X (gfx1250) — matrix operand order repaired and checked by reading, not compiled, not run on hardware
//
#include <hip/hip_runtime.h>
#include <hip/hip_bf16.h>
#include <stdint.h>

// ---------------------------------------------------------------------------
// Anderson-accelerated fixed point  x = x @ lin_w^T + lin_b  then final linear.
// MI455X strategy:
//  * history (X,Fm) kept in bf16: 168MB total -> resident in the 192MB L2
//  * GEMMs via v_wmma_f32_16x16x32_bf16 (fp32 accumulate)
//  * B panel staged to LDS by the Tensor Data Mover (tensor_load_to_lds with
//    per-row padding), fragments read back with ds_load_b128
//  * data-dependent while-loop -> 48 fixed iterations with device-side
//    `converged` early-exit flag (graph-capture safe, deterministic)
// ---------------------------------------------------------------------------

typedef __attribute__((ext_vector_type(16))) __bf16    v16bf;
typedef __attribute__((ext_vector_type(8)))  __bf16    v8bf;
typedef __attribute__((ext_vector_type(8)))  float     v8f;
typedef __attribute__((ext_vector_type(4)))  uint32_t  u32x4;
typedef __attribute__((ext_vector_type(8)))  uint32_t  u32x8;

#define R_ROWS 16384          // B*D*H = 16*8*128
#define NCOL   512
#define KDIM   512
#define BSZ    16
#define MHIST  5
#define NBATCH_ELEM (1024*512)        // per-batch elements inside a slot (2^19)
#define SLOT   (R_ROWS*NCOL)          // 8,388,608 elements per history slot
#define LAMBDA 1e-4f
#define TOLER  0.01f

// GEMM blocking
#define BM 128                // M rows per workgroup (8 waves x 16)
#define BN 64                 // N cols per workgroup (4 x 16 per wave)
#define KHALF 256             // K staged per TDM phase
#define LSTR  264             // LDS row stride in elems (256 + 8 pad = TDM pad 16B)

struct Ctrl {
  float gram[BSZ][15];     // upper triangle of 5x5 Gram, pair order i<=j
  float alpha[BSZ][MHIST];
  float num, den;          // residual accumulators
  int   converged;
  int   last_slot;
};

// ---------------------------------------------------------------------------
__global__ void cvt_f32_to_bf16(const float* __restrict__ src,
                                __bf16* __restrict__ dst, int n) {
  int i = blockIdx.x * blockDim.x + threadIdx.x;
  int stride = gridDim.x * blockDim.x;
  for (; i < n; i += stride) dst[i] = (__bf16)src[i];
}

__global__ void init_ctrl(Ctrl* c) {
  if (threadIdx.x == 0 && blockIdx.x == 0) {
    int* p = (int*)c;
    for (int i = 0; i < (int)(sizeof(Ctrl) / 4); ++i) p[i] = 0;
    c->last_slot = 1;   // matches (kf-1)%M if loop exited immediately
  }
}

// ---------------------------------------------------------------------------
// Gram: gram[b][p] = sum_n (Fm[m_i]-X[m_i])*(Fm[m_j]-X[m_j]) for pairs i<=j.
// Masking of inactive slots is applied later in the solve (only depends on k).
__global__ __launch_bounds__(256)
void gram_kernel(const __bf16* __restrict__ X, const __bf16* __restrict__ Fm,
                 Ctrl* __restrict__ c) {
  if (c->converged) return;
  const int b = blockIdx.y;
  const size_t base = (size_t)b * NBATCH_ELEM;
  float acc[15];
#pragma unroll
  for (int p = 0; p < 15; ++p) acc[p] = 0.f;

  const int tid  = blockIdx.x * blockDim.x + threadIdx.x;
  const int nthr = gridDim.x * blockDim.x;
  for (size_t e = (size_t)tid * 8; e < (size_t)NBATCH_ELEM; e += (size_t)nthr * 8) {
    float g[MHIST][8];
#pragma unroll
    for (int m = 0; m < MHIST; ++m) {
      const v8bf xv = *(const v8bf*)(X  + (size_t)m * SLOT + base + e);
      const v8bf fv = *(const v8bf*)(Fm + (size_t)m * SLOT + base + e);
#pragma unroll
      for (int q = 0; q < 8; ++q) g[m][q] = (float)fv[q] - (float)xv[q];
    }
    int p = 0;
#pragma unroll
    for (int i = 0; i < MHIST; ++i)
#pragma unroll
      for (int j = i; j < MHIST; ++j, ++p) {
        float s = 0.f;
#pragma unroll
        for (int q = 0; q < 8; ++q) s += g[i][q] * g[j][q];
        acc[p] += s;
      }
  }
  // wave32 reduction
#pragma unroll
  for (int p = 0; p < 15; ++p)
    for (int off = 16; off > 0; off >>= 1)
      acc[p] += __shfl_down(acc[p], off, 32);

  __shared__ float smem[8][15];
  const int lane = threadIdx.x & 31, wid = threadIdx.x >> 5;
  if (lane == 0)
#pragma unroll
    for (int p = 0; p < 15; ++p) smem[wid][p] = acc[p];
  __syncthreads();
  if (threadIdx.x < 15) {
    float s = 0.f;
#pragma unroll
    for (int w = 0; w < 8; ++w) s += smem[w][threadIdx.x];
    atomicAdd(&c->gram[b][threadIdx.x], s);
  }
}

// ---------------------------------------------------------------------------
// Per-batch 6x6 Gauss-Jordan solve of H * a = e0; alpha = a[1:6].
__global__ void solve_kernel(Ctrl* __restrict__ c, int k) {
  if (c->converged) return;
  const int b = threadIdx.x;
  if (b >= BSZ) return;
  const int nk = (k < MHIST) ? k : MHIST;

  float G[MHIST][MHIST];
  {
    int p = 0;
    for (int i = 0; i < MHIST; ++i)
      for (int j = i; j < MHIST; ++j, ++p) { G[i][j] = c->gram[b][p]; G[j][i] = G[i][j]; }
  }
  float H[6][7];
  for (int i = 0; i < 6; ++i)
    for (int j = 0; j < 7; ++j) H[i][j] = 0.f;
  for (int j = 0; j < MHIST; ++j) {
    const float mj = (j < nk) ? 1.f : 0.f;
    H[0][j + 1] = mj; H[j + 1][0] = mj;
  }
  for (int i = 0; i < MHIST; ++i)
    for (int j = 0; j < MHIST; ++j) {
      const float mi = (i < nk) ? 1.f : 0.f;
      const float mj = (j < nk) ? 1.f : 0.f;
      H[i + 1][j + 1] = mi * mj * G[i][j] + ((i == j) ? LAMBDA : 0.f);
    }
  H[0][6] = 1.f;  // rhs y = e0

  for (int col = 0; col < 6; ++col) {
    int piv = col; float best = fabsf(H[col][col]);
    for (int r = col + 1; r < 6; ++r) {
      const float v = fabsf(H[r][col]);
      if (v > best) { best = v; piv = r; }
    }
    if (piv != col)
      for (int j = 0; j < 7; ++j) { const float t = H[col][j]; H[col][j] = H[piv][j]; H[piv][j] = t; }
    const float inv = 1.f / H[col][col];
    for (int r = 0; r < 6; ++r)
      if (r != col) {
        const float f = H[r][col] * inv;
        for (int j = col; j < 7; ++j) H[r][j] -= f * H[col][j];
      }
  }
  for (int m = 0; m < MHIST; ++m)
    c->alpha[b][m] = H[m + 1][6] / H[m + 1][m + 1];
  // reset gram accumulators for the next iteration
  for (int p = 0; p < 15; ++p) c->gram[b][p] = 0.f;
}

// ---------------------------------------------------------------------------
// xnew = sum_m alpha[b][m] * Fm[m]   (BETA == 1)   -> written into X slot idx
__global__ __launch_bounds__(256)
void xnew_kernel(const __bf16* __restrict__ Fm, __bf16* __restrict__ Xs,
                 const Ctrl* __restrict__ c) {
  if (c->converged) return;
  const size_t i = ((size_t)blockIdx.x * blockDim.x + threadIdx.x) * 8;
  if (i >= (size_t)SLOT) return;
  const int b = (int)(i >> 19);
  float a[MHIST];
#pragma unroll
  for (int m = 0; m < MHIST; ++m) a[m] = c->alpha[b][m];
  float acc[8];
#pragma unroll
  for (int q = 0; q < 8; ++q) acc[q] = 0.f;
#pragma unroll
  for (int m = 0; m < MHIST; ++m) {
    const v8bf f = *(const v8bf*)(Fm + (size_t)m * SLOT + i);
#pragma unroll
    for (int q = 0; q < 8; ++q) acc[q] += a[m] * (float)f[q];
  }
  v8bf o;
#pragma unroll
  for (int q = 0; q < 8; ++q) o[q] = (__bf16)acc[q];
  *(v8bf*)(Xs + i) = o;
}

// ---------------------------------------------------------------------------
// D = A @ B^T + bias  (rows=16384, N=K=512) via v_wmma_f32_16x16x32_bf16.
// Workgroup tile 128x64: 8 waves, each wave = one 16-row M-tile x four N-tiles.
// B panel (BN x KHALF bf16) staged to LDS by the Tensor Data Mover with 16B
// per-row padding (LDS stride 264 elems -> bank-friendly ds_load_b128 reads).
//
// NOTE: the LDS base address is derived from the real `lb` pointer and fed
// into the descriptor. This (a) fills D#.lds_addr correctly and (b) escapes
// the shared array into the asm, so the "memory" clobber makes the compiler
// treat the TDM as writing lb (otherwise the ds_load fragments fold to undef).
__device__ __forceinline__ void tdm_stage_b_panel(__bf16* lds, const __bf16* gsrc,
                                                  int kphase) {
  const uint32_t lds_addr = (uint32_t)(uintptr_t)lds;   // low 32 bits = LDS offset
  const uint64_t ga = (uint64_t)(const void*)(gsrc + kphase * KHALF);
  // D# group0: count=1 | lds_addr | global_addr | type=2 ("image")
  u32x4 g0;
  g0.x = 1u;
  g0.y = lds_addr;
  g0.z = (uint32_t)ga;
  g0.w = (uint32_t)((ga >> 32) & 0x01FFFFFFu) | (2u << 30);
  // D# group1: data_size=2B, pad every 128 DWORDs (512B row) by 4 DWORDs (16B)
  u32x8 g1;
  g1[0] = (1u << 16) | (1u << 20) | (6u << 22) | (3u << 25);
  g1[1] = ((uint32_t)KDIM & 0xFFFFu) << 16;            // tensor_dim0 lo
  g1[2] = (((uint32_t)KDIM >> 16) & 0xFFFFu)           // tensor_dim0 hi
        | (((uint32_t)KDIM & 0xFFFFu) << 16);          // tensor_dim1 lo (512 rows)
  g1[3] = (((uint32_t)KDIM >> 16) & 0xFFFFu)           // tensor_dim1 hi
        | ((uint32_t)KHALF << 16);                     // tile_dim0 = 256
  g1[4] = (uint32_t)BN;                                // tile_dim1 = 64, tile_dim2 = 0
  g1[5] = (uint32_t)KDIM;                              // tensor_dim0_stride lo32 = 512
  g1[6] = 0u;                                          // stride hi | dim1_stride lo
  g1[7] = 0u;
  asm volatile("tensor_load_to_lds %0, %1" :: "s"(g0), "s"(g1) : "memory");
  __builtin_amdgcn_s_wait_tensorcnt(0);
}

__global__ __launch_bounds__(256)
void gemm_bf16_kernel(const __bf16* __restrict__ Abase,
                      const __bf16* __restrict__ B,
                      const float*  __restrict__ bias,
                      __bf16* __restrict__ Dbf,     // nullable
                      float*  __restrict__ Df32,    // nullable
                      const __bf16* __restrict__ Xres,  // nullable
                      Ctrl* __restrict__ c,
                      int check_conv, int use_last_slot) {
  __shared__ __bf16 lb[BN * LSTR];   // 33,792 B staged B panel
  if (check_conv && c->converged) return;   // uniform across the block

  const __bf16* A = Abase;
  if (use_last_slot) A += (size_t)c->last_slot * SLOT;

  const int bm = blockIdx.x >> 3;          // 128 M panels
  const int bn = blockIdx.x & 7;           // 8  N panels
  const int m_base = bm * BM;
  const int n_base = bn * BN;

  const int lane = threadIdx.x & 31;
  const int wid  = threadIdx.x >> 5;
  const int half = lane >> 4;              // K-half selector (contiguous k = 16*half + s)
  const int l    = lane & 15;
  const int m0   = m_base + wid * 16;

  const __bf16* arow   = A + (size_t)(m0 + l) * KDIM + half * 16;
  const __bf16* bpanel = B + (size_t)n_base * KDIM;

  v8f acc0 = {}, acc1 = {}, acc2 = {}, acc3 = {};

#pragma unroll
  for (int ph = 0; ph < 2; ++ph) {
    __syncthreads();                       // previous phase fully consumed
    if (wid == 0) tdm_stage_b_panel(lb, bpanel, ph);  // TDM ignores EXEC; wave0 issues
    __syncthreads();

    const int kglob = ph * KHALF;
    for (int kk = 0; kk < KHALF; kk += 32) {
      union { v16bf v; v8bf h[2]; } ua;
      ua.h[0] = *(const v8bf*)(arow + kglob + kk);
      ua.h[1] = *(const v8bf*)(arow + kglob + kk + 8);
      __builtin_prefetch(arow + kglob + kk + 32, 0, 1);

      const int kb = kk + half * 16;
      const v16bf b0 = *(const v16bf*)(lb + (size_t)(l)      * LSTR + kb);
      const v16bf b1 = *(const v16bf*)(lb + (size_t)(16 + l) * LSTR + kb);
      const v16bf b2 = *(const v16bf*)(lb + (size_t)(32 + l) * LSTR + kb);
      const v16bf b3 = *(const v16bf*)(lb + (size_t)(48 + l) * LSTR + kb);

      acc0 = __builtin_amdgcn_wmma_f32_16x16x32_bf16(false, ua.v, false, b0,
                                                     (short)0, acc0, false, false);
      acc1 = __builtin_amdgcn_wmma_f32_16x16x32_bf16(false, ua.v, false, b1,
                                                     (short)0, acc1, false, false);
      acc2 = __builtin_amdgcn_wmma_f32_16x16x32_bf16(false, ua.v, false, b2,
                                                     (short)0, acc2, false, false);
      acc3 = __builtin_amdgcn_wmma_f32_16x16x32_bf16(false, ua.v, false, b3,
                                                     (short)0, acc3, false, false);
    }
  }

  // ---- epilogue: bias add, optional bf16/f32 stores, fused residual ----
  float rn = 0.f, rd = 0.f;
  v8f accs[4] = {acc0, acc1, acc2, acc3};
#pragma unroll
  for (int t = 0; t < 4; ++t) {
    const int col = n_base + t * 16 + l;
    const float bb = bias[col];
#pragma unroll
    for (int r = 0; r < 8; ++r) {
      const int row = m0 + half * 8 + r;          // C/D VGPR layout per ISA
      const float d = accs[t][r] + bb;
      const size_t off = (size_t)row * NCOL + col;
      if (Dbf)  Dbf[off]  = (__bf16)d;
      if (Df32) Df32[off] = d;
      if (Xres) {
        const float xr = (float)Xres[off];
        const float df = d - xr;
        rn += df * df; rd += d * d;
      }
    }
  }
  if (Xres) {
    for (int off = 16; off > 0; off >>= 1) {
      rn += __shfl_down(rn, off, 32);
      rd += __shfl_down(rd, off, 32);
    }
    if (lane == 0) { atomicAdd(&c->num, rn); atomicAdd(&c->den, rd); }
  }
}

// ---------------------------------------------------------------------------
__global__ void update_kernel(Ctrl* c, int idx) {
  if (threadIdx.x == 0 && blockIdx.x == 0) {
    if (!c->converged) {
      c->last_slot = idx;
      const float res = sqrtf(c->num) / (1e-5f + sqrtf(c->den));
      if (res < TOLER) c->converged = 1;
    }
    c->num = 0.f; c->den = 0.f;
  }
}

// ---------------------------------------------------------------------------
extern "C" void kernel_launch(void* const* d_in, const int* in_sizes, int n_in,
                              void* d_out, int out_size, void* d_ws, size_t ws_size,
                              hipStream_t stream) {
  const float* x      = (const float*)d_in[0];   // (16,8,128,512)
  const float* lin_w  = (const float*)d_in[1];   // (512,512)
  const float* lin_b  = (const float*)d_in[2];   // (512,)
  const float* weight = (const float*)d_in[3];   // (512,512)
  const float* bias   = (const float*)d_in[4];   // (512,)
  float* out = (float*)d_out;

  __bf16* Xh = (__bf16*)d_ws;                         // 5 slots
  __bf16* Fh = Xh + (size_t)MHIST * SLOT;             // 5 slots
  __bf16* WL = Fh + (size_t)MHIST * SLOT;             // lin_w bf16
  __bf16* WO = WL + (size_t)KDIM * NCOL;              // weight bf16
  Ctrl* ctrl = (Ctrl*)(((uintptr_t)(WO + (size_t)KDIM * NCOL) + 255) & ~(uintptr_t)255);

  const dim3 blk(256);
  const int GEMM_GRID = (R_ROWS / BM) * (NCOL / BN);   // 128 * 8 = 1024 blocks

  // ---- setup ----
  cvt_f32_to_bf16<<<4096, blk, 0, stream>>>(x, Xh, SLOT);          // X slot0 = x0
  cvt_f32_to_bf16<<<1024, blk, 0, stream>>>(lin_w, WL, KDIM * NCOL);
  cvt_f32_to_bf16<<<1024, blk, 0, stream>>>(weight, WO, KDIM * NCOL);
  init_ctrl<<<1, 1, 0, stream>>>(ctrl);

  // F0 = f(x0) -> Fm slot0
  gemm_bf16_kernel<<<GEMM_GRID, blk, 0, stream>>>(Xh, WL, lin_b,
      Fh, nullptr, nullptr, ctrl, 0, 0);
  // X slot1 = F0
  hipMemcpyAsync(Xh + (size_t)SLOT, Fh, (size_t)SLOT * sizeof(__bf16),
                 hipMemcpyDeviceToDevice, stream);
  // Fm slot1 = f(F0)
  gemm_bf16_kernel<<<GEMM_GRID, blk, 0, stream>>>(Xh + (size_t)SLOT, WL, lin_b,
      Fh + (size_t)SLOT, nullptr, nullptr, ctrl, 0, 0);

  // ---- Anderson loop: k = 2..49, slot idx = k % 5 (k advances deterministically)
  for (int k = 2; k < 50; ++k) {
    const int idx = k % MHIST;
    gram_kernel<<<dim3(64, BSZ), blk, 0, stream>>>(Xh, Fh, ctrl);
    solve_kernel<<<1, 32, 0, stream>>>(ctrl, k);
    xnew_kernel<<<4096, blk, 0, stream>>>(Fh, Xh + (size_t)idx * SLOT, ctrl);
    // fnew = f(xnew) -> Fm slot idx, with fused residual vs xnew
    gemm_bf16_kernel<<<GEMM_GRID, blk, 0, stream>>>(Xh + (size_t)idx * SLOT, WL, lin_b,
        Fh + (size_t)idx * SLOT, nullptr, Xh + (size_t)idx * SLOT, ctrl, 1, 0);
    update_kernel<<<1, 1, 0, stream>>>(ctrl, idx);
  }

  // ---- final: out = X[last_slot] @ weight^T + bias  (fp32 output)
  gemm_bf16_kernel<<<GEMM_GRID, blk, 0, stream>>>(Xh, WO, bias,
      nullptr, out, nullptr, ctrl, 0, 1);

  (void)in_sizes; (void)n_in; (void)out_size; (void)ws_size;
}
